// SingleHead_33646773797571
// MI455X (gfx1250) — matrix operand order repaired and checked by reading, not compiled, not run on hardware
//
#include <hip/hip_runtime.h>

// Problem constants (from reference): B=8, T=2048, C=1024, D=64
#define BATCH 8
#define SEQ   2048
#define CDIM  1024
#define DDIM  64

typedef __attribute__((ext_vector_type(16))) __bf16 v16bf;
typedef __attribute__((ext_vector_type(8)))  float  v8f;

union Frag {
    v16bf v;
    uint4 q[2];
    unsigned short h[16];
};

// round-to-nearest-even fp32 -> bf16
static __device__ __forceinline__ unsigned short f2bf(float f) {
    union { float f; unsigned int u; } x; x.f = f;
    unsigned int r = x.u + 0x7fffu + ((x.u >> 16) & 1u);
    return (unsigned short)(r >> 16);
}

static __device__ __forceinline__ unsigned int pack2bf(float lo, float hi) {
    return ((unsigned int)f2bf(hi) << 16) | (unsigned int)f2bf(lo);
}

// ---------------------------------------------------------------------------
// Kernel 1: convert Wq/Wk/Wv fp32 [C,D] -> bf16 transposed Wt [3][D][C]
// so WMMA B-fragments (lane = out column, K contiguous) load 32B per lane.
// ---------------------------------------------------------------------------
__global__ void prep_weights(const float* __restrict__ Wq,
                             const float* __restrict__ Wk,
                             const float* __restrict__ Wv,
                             unsigned short* __restrict__ Wt) {
    int idx = blockIdx.x * blockDim.x + threadIdx.x;   // 0 .. 3*C*D-1
    int m   = idx >> 16;                               // which matrix (C*D = 65536)
    int rem = idx & 65535;
    int k   = rem >> 6;                                // row in C
    int n   = rem & 63;                                // col in D
    const float* W = (m == 0) ? Wq : (m == 1) ? Wk : Wv;
    Wt[m * (DDIM * CDIM) + n * CDIM + k] = f2bf(W[rem]);
}

// ---------------------------------------------------------------------------
// Kernel 2: QKV projection.  Grid = (B*T)/128 blocks, 256 thr = 8 waves.
// Each wave: one 16-row strip, 12 n-tiles (Q:4, K:4, V:4), K-loop over C.
// x staged fp32->bf16 in double-buffered LDS (float4 loads, b64 packed
// stores); all 12 B-fragments preloaded per k-step so loads overlap.
// Q scaled by 1/sqrt(D); V stored transposed [B][D][T] bf16.
// ---------------------------------------------------------------------------
__global__ void __launch_bounds__(256)
qkv_proj(const float* __restrict__ x, const unsigned short* __restrict__ Wt,
         unsigned short* __restrict__ Qb, unsigned short* __restrict__ Kb,
         unsigned short* __restrict__ Vt) {
    __shared__ unsigned short xs[2][128 * 32];   // 2 x 8 KB bf16 tiles of x
    const int tid  = threadIdx.x;
    const int wave = tid >> 5, lane = tid & 31;
    const int ln = lane & 15, hal = lane >> 4;
    const int row0 = blockIdx.x * 128;           // flattened [B*T] row base

    v8f acc[12];
    #pragma unroll
    for (int t = 0; t < 12; ++t) acc[t] = (v8f){};

    for (int k0 = 0; k0 < CDIM; k0 += 32) {
        const int buf = (k0 >> 5) & 1;
        // cooperative staging: 1024 float4 loads across 256 threads
        #pragma unroll
        for (int i = 0; i < 4; ++i) {
            int idx4 = i * 256 + tid;            // float4 index in 128x32 tile
            int r = idx4 >> 3, c4 = idx4 & 7;
            float4 f = *(const float4*)&x[(size_t)(row0 + r) * CDIM + k0 + c4 * 4];
            *(uint2*)&xs[buf][r * 32 + c4 * 4] =
                make_uint2(pack2bf(f.x, f.y), pack2bf(f.z, f.w));
        }
        __syncthreads();

        // preload all 12 B-fragments (overlapping loads)
        Frag bm[12];
        #pragma unroll
        for (int t = 0; t < 12; ++t) {
            const int m = t >> 2, n0 = (t & 3) * 16;
            const unsigned short* wp =
                Wt + m * (DDIM * CDIM) + (n0 + ln) * CDIM + k0 + hal * 16;
            bm[t].q[0] = *(const uint4*)(wp);
            bm[t].q[1] = *(const uint4*)(wp + 8);
        }

        // A fragment: 16x32 bf16, lane<16 K{0..15}, lane>=16 K{16..31}
        Frag a;
        const unsigned short* ap = &xs[buf][(wave * 16 + ln) * 32 + hal * 16];
        a.q[0] = *(const uint4*)(ap);
        a.q[1] = *(const uint4*)(ap + 8);

        #pragma unroll
        for (int t = 0; t < 12; ++t)
            acc[t] = __builtin_amdgcn_wmma_f32_16x16x32_bf16(
                false, a.v, false, bm[t].v, (short)0, acc[t], false, false);
        // no trailing barrier: next iteration writes the other xs buffer
    }

    const int bidx = row0 >> 11;   // batch (T = 2048)
    #pragma unroll
    for (int t = 0; t < 12; ++t) {
        const int m = t >> 2, n0 = (t & 3) * 16;
        const int col = n0 + ln;
        #pragma unroll
        for (int r = 0; r < 8; ++r) {
            int row = row0 + wave * 16 + r + hal * 8;   // C-layout row
            float vv = acc[t][r];
            if (m == 0)      Qb[(size_t)row * DDIM + col] = f2bf(vv * 0.125f);
            else if (m == 1) Kb[(size_t)row * DDIM + col] = f2bf(vv);
            else             Vt[((size_t)bidx * DDIM + col) * SEQ + (row - bidx * SEQ)]
                                 = f2bf(vv);
        }
    }
}

// ---------------------------------------------------------------------------
// Kernel 3: fused flash attention (causal, online softmax).
// Grid = (T/128, B), 256 thr = 8 waves; wave owns a 16-row q strip.
// K-fragments preloaded in groups of 4 n-tiles; V-fragments preloaded per
// K-chunk; P staged bf16 through double-buffered per-wave LDS (one barrier
// per k-tile); next K/V tiles prefetched during softmax.
// ---------------------------------------------------------------------------
__global__ void __launch_bounds__(256)
flash_attn(const unsigned short* __restrict__ Qb,
           const unsigned short* __restrict__ Kb,
           const unsigned short* __restrict__ Vt,
           float* __restrict__ out) {
    __shared__ unsigned short pbuf[2][8][16 * 128];   // 2 x 32 KB P staging
    const int tid  = threadIdx.x;
    const int wave = tid >> 5, lane = tid & 31;
    const int ln = lane & 15, hal = lane >> 4;
    const int qt = blockIdx.x, b = blockIdx.y;
    const int q0 = qt * 128;

    // Q fragments for this wave's 16 rows (D=64 -> two K=32 fragments)
    const unsigned short* qp =
        Qb + ((size_t)b * SEQ + q0 + wave * 16 + ln) * DDIM;
    Frag aq0, aq1;
    aq0.q[0] = *(const uint4*)(qp + hal * 16);
    aq0.q[1] = *(const uint4*)(qp + 8 + hal * 16);
    aq1.q[0] = *(const uint4*)(qp + 32 + hal * 16);
    aq1.q[1] = *(const uint4*)(qp + 40 + hal * 16);

    float mrow[8], lrow[8];
    v8f oacc[4];
    #pragma unroll
    for (int r = 0; r < 8; ++r) { mrow[r] = -1e30f; lrow[r] = 0.0f; }
    #pragma unroll
    for (int d = 0; d < 4; ++d) oacc[d] = (v8f){};

    for (int kt = 0; kt <= qt; ++kt) {
        const int k0 = kt * 128;
        const int pb = kt & 1;

        // ---- S = Q K^T (8 tiles of 16 keys, fragment loads in groups of 4) ----
        v8f s[8];
        #pragma unroll
        for (int g = 0; g < 2; ++g) {
            Frag bk[4][2];
            #pragma unroll
            for (int j = 0; j < 4; ++j) {
                const int nt = g * 4 + j;
                const unsigned short* kp =
                    Kb + ((size_t)b * SEQ + k0 + nt * 16 + ln) * DDIM + hal * 16;
                bk[j][0].q[0] = *(const uint4*)(kp);
                bk[j][0].q[1] = *(const uint4*)(kp + 8);
                bk[j][1].q[0] = *(const uint4*)(kp + 32);
                bk[j][1].q[1] = *(const uint4*)(kp + 40);
            }
            #pragma unroll
            for (int j = 0; j < 4; ++j) {
                v8f z = (v8f){};
                z = __builtin_amdgcn_wmma_f32_16x16x32_bf16(
                    false, aq0.v, false, bk[j][0].v, (short)0, z, false, false);
                z = __builtin_amdgcn_wmma_f32_16x16x32_bf16(
                    false, aq1.v, false, bk[j][1].v, (short)0, z, false, false);
                s[g * 4 + j] = z;
            }
        }

        // prefetch next k-tile of K and V while we do softmax VALU work
        if (kt < qt) {
            const int kn = k0 + 128;
            __builtin_prefetch(Kb + ((size_t)b * SEQ + kn + lane * 4) * DDIM, 0, 3);
            __builtin_prefetch(
                Vt + ((size_t)b * DDIM + lane * 2) * SEQ + kn, 0, 3);
        }

        // ---- causal mask on the diagonal block ----
        if (kt == qt) {
            #pragma unroll
            for (int nt = 0; nt < 8; ++nt)
                #pragma unroll
                for (int r = 0; r < 8; ++r) {
                    int col = k0 + nt * 16 + ln;
                    int row = q0 + wave * 16 + r + hal * 8;
                    if (col > row) s[nt][r] = -1e30f;
                }
        }

        // ---- online softmax: row max, rescale, exp, row sum ----
        float mnew[8], scalef[8];
        #pragma unroll
        for (int r = 0; r < 8; ++r) {
            float mt = s[0][r];
            #pragma unroll
            for (int nt = 1; nt < 8; ++nt) mt = fmaxf(mt, s[nt][r]);
            #pragma unroll
            for (int off = 1; off < 16; off <<= 1)
                mt = fmaxf(mt, __shfl_xor(mt, off, 32));
            mnew[r]   = fmaxf(mrow[r], mt);
            scalef[r] = __expf(mrow[r] - mnew[r]);
            lrow[r]  *= scalef[r];
            mrow[r]   = mnew[r];
        }
        #pragma unroll
        for (int d = 0; d < 4; ++d)
            #pragma unroll
            for (int r = 0; r < 8; ++r) oacc[d][r] *= scalef[r];

        float ladd[8];
        #pragma unroll
        for (int r = 0; r < 8; ++r) ladd[r] = 0.0f;
        #pragma unroll
        for (int nt = 0; nt < 8; ++nt)
            #pragma unroll
            for (int r = 0; r < 8; ++r) {
                float p = __expf(s[nt][r] - mnew[r]);
                ladd[r] += p;
                pbuf[pb][wave][(r + hal * 8) * 128 + nt * 16 + ln] = f2bf(p);
            }
        #pragma unroll
        for (int r = 0; r < 8; ++r) {
            float t = ladd[r];
            #pragma unroll
            for (int off = 1; off < 16; off <<= 1) t += __shfl_xor(t, off, 32);
            lrow[r] += t;
        }
        __syncthreads();   // P visible across lanes; double buffer -> 1 barrier

        // ---- O += P V  (4 K-chunks of 32, 4 d-tiles of 16) ----
        #pragma unroll
        for (int kc = 0; kc < 4; ++kc) {
            Frag apf;
            const unsigned short* pp =
                &pbuf[pb][wave][ln * 128 + kc * 32 + hal * 16];
            apf.q[0] = *(const uint4*)(pp);
            apf.q[1] = *(const uint4*)(pp + 8);
            Frag bv[4];
            #pragma unroll
            for (int d = 0; d < 4; ++d) {
                const unsigned short* vp =
                    Vt + ((size_t)b * DDIM + d * 16 + ln) * SEQ + k0 + kc * 32 + hal * 16;
                bv[d].q[0] = *(const uint4*)(vp);
                bv[d].q[1] = *(const uint4*)(vp + 8);
            }
            #pragma unroll
            for (int d = 0; d < 4; ++d)
                oacc[d] = __builtin_amdgcn_wmma_f32_16x16x32_bf16(
                    false, apf.v, false, bv[d].v, (short)0, oacc[d], false, false);
        }
    }

    // ---- normalize + store fp32 ----
    #pragma unroll
    for (int d = 0; d < 4; ++d)
        #pragma unroll
        for (int r = 0; r < 8; ++r) {
            int row = q0 + wave * 16 + r + hal * 8;
            out[((size_t)b * SEQ + row) * DDIM + d * 16 + ln] = oacc[d][r] / lrow[r];
        }
}

// ---------------------------------------------------------------------------
extern "C" void kernel_launch(void* const* d_in, const int* in_sizes, int n_in,
                              void* d_out, int out_size, void* d_ws, size_t ws_size,
                              hipStream_t stream) {
    const float* x  = (const float*)d_in[0];
    const float* Wq = (const float*)d_in[1];
    const float* Wk = (const float*)d_in[2];
    const float* Wv = (const float*)d_in[3];

    unsigned short* ws = (unsigned short*)d_ws;
    unsigned short* Wt = ws;                                   // 3*64*1024
    unsigned short* Qb = Wt + 3 * DDIM * CDIM;                 // B*T*64
    unsigned short* Kb = Qb + (size_t)BATCH * SEQ * DDIM;
    unsigned short* Vt = Kb + (size_t)BATCH * SEQ * DDIM;      // [B][64][T]

    prep_weights<<<(3 * CDIM * DDIM) / 256, 256, 0, stream>>>(Wq, Wk, Wv, Wt);
    qkv_proj<<<(BATCH * SEQ) / 128, 256, 0, stream>>>(x, Wt, Qb, Kb, Vt);
    dim3 g(SEQ / 128, BATCH);
    flash_attn<<<g, 256, 0, stream>>>(Qb, Kb, Vt, (float*)d_out);
}
